// ABSEncoder_51926154608749
// MI455X (gfx1250) — matrix operand order repaired and checked by reading, not compiled, not run on hardware
//
#include <hip/hip_runtime.h>

#define BB 16
#define SS 1024
#define DD 1024
#define CC 4096
#define NEGV (-1.0e9f)

// Flip to 0 if the assembler rejects the async-LDS mnemonic on this toolchain.
#define USE_ASYNC_LDS 1

typedef __bf16 bf16_t;
typedef __attribute__((ext_vector_type(8)))  __bf16 v8bf;
typedef __attribute__((ext_vector_type(16))) __bf16 v16bf;
typedef __attribute__((ext_vector_type(8)))  float  v8f;

static __device__ inline v16bf cat16(v8bf lo, v8bf hi) {
  v16bf r;
#pragma unroll
  for (int i = 0; i < 8; ++i) { r[i] = lo[i]; r[8 + i] = hi[i]; }
  return r;
}

// A fragment (16x32 bf16, row-major source). row ptr points at [m][k0].
static __device__ inline v16bf load_a_frag(const bf16_t* row_k0, int hi) {
  v8bf lo = *reinterpret_cast<const v8bf*>(row_k0 + 8 * hi);
  v8bf hh = *reinterpret_cast<const v8bf*>(row_k0 + 16 + 8 * hi);
  return cat16(lo, hh);
}

// B fragment (32x16 bf16, column-major source: one column contiguous over K).
static __device__ inline v16bf load_b_frag(const bf16_t* col_k0, int hi) {
  v8bf lo = *reinterpret_cast<const v8bf*>(col_k0 + 8 * hi);
  v8bf hh = *reinterpret_cast<const v8bf*>(col_k0 + 16 + 8 * hi);
  return cat16(lo, hh);
}

static __device__ inline v8f wmma_bf16(v16bf a, v16bf b, v8f c) {
  return __builtin_amdgcn_wmma_f32_16x16x32_bf16(false, a, false, b, (short)0, c,
                                                 false, false);
}

// ---- CDNA5 async memory->LDS staging (ASYNCcnt path) ----
static __device__ inline void stage16(const bf16_t* gsrc, bf16_t* lds_dst) {
#if USE_ASYNC_LDS
  unsigned int laddr = (unsigned int)(uintptr_t)lds_dst;  // aperture: addr[31:0] = LDS offset
  asm volatile("global_load_async_to_lds_b128 %0, %1, off"
               :: "v"(laddr), "v"(gsrc) : "memory");
#else
  *reinterpret_cast<v8bf*>(lds_dst) = *reinterpret_cast<const v8bf*>(gsrc);
#endif
}
static __device__ inline void stage_wait() {
#if USE_ASYNC_LDS
  asm volatile("s_wait_asynccnt 0" ::: "memory");
#endif
}

// ---------------- gather / convert kernels ----------------

__global__ void gather_x_kernel(const int* __restrict__ x, const float* __restrict__ Femb,
                                bf16_t* __restrict__ xbf, bf16_t* __restrict__ xtbf) {
  int gid = blockIdx.x * blockDim.x + threadIdx.x;   // one thread per 8 dims
  int d8 = gid & (DD / 8 - 1);
  int rs = gid >> 7;
  int s  = rs & (SS - 1);
  int b  = rs >> 10;
  int tok = x[b * SS + s];
  const float* src = Femb + (size_t)tok * DD + d8 * 8;
  v8bf v;
#pragma unroll
  for (int i = 0; i < 8; ++i) v[i] = (bf16_t)src[i];
  *reinterpret_cast<v8bf*>(xbf + ((size_t)(b * SS + s)) * DD + d8 * 8) = v;
  bf16_t* xt = xtbf + (size_t)b * DD * SS;
#pragma unroll
  for (int i = 0; i < 8; ++i) xt[(size_t)(d8 * 8 + i) * SS + s] = v[i];
}

__global__ void gather_yc_kernel(const int* __restrict__ yc, const float* __restrict__ Gemb,
                                 bf16_t* __restrict__ ycr) {
  int gid = blockIdx.x * blockDim.x + threadIdx.x;   // one thread per 8 dims
  int d8 = gid & 127;
  int bt = gid >> 7;
  int t  = bt & 4095;
  int b  = bt >> 12;
  int tok = yc[b * 4096 + t];
  const float* src = Gemb + (size_t)tok * DD + d8 * 8;
  v8bf v;
#pragma unroll
  for (int i = 0; i < 8; ++i) v[i] = (bf16_t)src[i];
  *reinterpret_cast<v8bf*>(ycr + ((size_t)b * 4096 + t) * DD + d8 * 8) = v;
}

__global__ void pwt_kernel(const float* __restrict__ Pw, bf16_t* __restrict__ pwt) {
  int gid = blockIdx.x * blockDim.x + threadIdx.x;   // total DD*CC/8
  int c8 = gid & 511;
  int j  = gid >> 9;
  v8bf v;
#pragma unroll
  for (int i = 0; i < 8; ++i) v[i] = (bf16_t)Pw[(size_t)(c8 * 8 + i) * DD + j];
  *reinterpret_cast<v8bf*>(pwt + (size_t)j * CC + c8 * 8) = v;
}

// ---------------- GEMM 1: p_y (stored transposed, bf16, bias fused) ----------------
// mt[b][j][i] = bf16( sum_c ycr[b][i][c] * P_w[c][j] + P_b[j] )
__global__ void pygemm_kernel(const bf16_t* __restrict__ ycr, const bf16_t* __restrict__ pwt,
                              const float* __restrict__ Pb, bf16_t* __restrict__ mt) {
  extern __shared__ char smem[];
  bf16_t* As = (bf16_t*)smem;                 // 16 rows x 4096 = 128 KB
  int b  = blockIdx.x >> 6;
  int rb = blockIdx.x & 63;
  int t  = threadIdx.x;

  const bf16_t* asrc = ycr + ((size_t)b * 1024 + rb * 16) * CC;
#pragma unroll 4
  for (int i = 0; i < 64; ++i) {
    int e = (i * 128 + t) * 8;
    stage16(asrc + e, As + e);
  }
  stage_wait();
  __syncthreads();

  int wave = t >> 5, lane = t & 31;
  int hi = lane >> 4, lo16 = lane & 15;
  const bf16_t* arow = As + lo16 * CC;

  for (int g = 0; g < 4; ++g) {
    int ct0 = g * 16 + wave * 4;              // 4 adjacent tiles per wave
    const bf16_t* bc0 = pwt + (size_t)(ct0 * 16 + lo16) * CC;
    v8f a0 = {}, a1 = {}, a2 = {}, a3 = {};
    for (int k0 = 0; k0 < CC; k0 += 32) {
      v16bf af = load_a_frag(arow + k0, hi);
      v16bf b0 = load_b_frag(bc0 + k0, hi);
      v16bf b1 = load_b_frag(bc0 + 16 * CC + k0, hi);
      v16bf b2 = load_b_frag(bc0 + 32 * CC + k0, hi);
      v16bf b3 = load_b_frag(bc0 + 48 * CC + k0, hi);
      __builtin_prefetch(bc0 + k0 + 256, 0, 1);
      a0 = wmma_bf16(af, b0, a0);
      a1 = wmma_bf16(af, b1, a1);
      a2 = wmma_bf16(af, b2, a2);
      a3 = wmma_bf16(af, b3, a3);
    }
#pragma unroll
    for (int j = 0; j < 4; ++j) {
      v8f acc = (j == 0) ? a0 : (j == 1) ? a1 : (j == 2) ? a2 : a3;
      int col = (ct0 + j) * 16 + lo16;
      float bias = Pb[col];
      bf16_t* dst = mt + ((size_t)b * 1024 + col) * 1024 + rb * 16 + 8 * hi;
      v8bf o;
#pragma unroll
      for (int r = 0; r < 8; ++r) o[r] = (bf16_t)(acc[r] + bias);
      *reinterpret_cast<v8bf*>(dst) = o;
    }
  }
}

// ---------------- fused scores + softmax + output GEMM ----------------
__global__ void attn_kernel(const bf16_t* __restrict__ xbf, const bf16_t* __restrict__ xtbf,
                            const bf16_t* __restrict__ mt, const int* __restrict__ mask,
                            float* __restrict__ out) {
  extern __shared__ char smem[];
  bf16_t* xs  = (bf16_t*)smem;                           // 16 x 1024 bf16 = 32 KB
  float*  sc  = (float*)(smem + 32768);                  // 16 x 1024 f32  = 64 KB
  bf16_t* ab  = (bf16_t*)(smem + 32768 + 65536);         // 16 x 1024 bf16 = 32 KB
  float*  red = (float*)(smem + 32768 + 65536 + 32768);  // 16 x 8 partials

  int b  = blockIdx.x >> 6;
  int rb = blockIdx.x & 63;
  int t  = threadIdx.x;

  const bf16_t* xsrc = xbf + ((size_t)b * SS + rb * 16) * DD;
#pragma unroll 4
  for (int i = 0; i < 16; ++i) {
    int e = (i * 128 + t) * 8;
    stage16(xsrc + e, xs + e);
  }
  stage_wait();
  __syncthreads();

  int wave = t >> 5, lane = t & 31;
  int hi = lane >> 4, lo16 = lane & 15;

  // Phase 1: scores = x_e @ p_y + mask*NEG  (into LDS, f32)
  {
    const bf16_t* arow = xs + lo16 * DD;
    for (int g = 0; g < 4; ++g) {
      int ct0 = g * 16 + wave * 4;
      const bf16_t* bc0 = mt + ((size_t)b * 1024 + ct0 * 16 + lo16) * 1024;
      v8f a0 = {}, a1 = {}, a2 = {}, a3 = {};
      for (int k0 = 0; k0 < DD; k0 += 32) {
        v16bf af = load_a_frag(arow + k0, hi);
        v16bf b0 = load_b_frag(bc0 + k0, hi);
        v16bf b1 = load_b_frag(bc0 + 16 * 1024 + k0, hi);
        v16bf b2 = load_b_frag(bc0 + 32 * 1024 + k0, hi);
        v16bf b3 = load_b_frag(bc0 + 48 * 1024 + k0, hi);
        __builtin_prefetch(bc0 + k0 + 256, 0, 1);
        a0 = wmma_bf16(af, b0, a0);
        a1 = wmma_bf16(af, b1, a1);
        a2 = wmma_bf16(af, b2, a2);
        a3 = wmma_bf16(af, b3, a3);
      }
#pragma unroll
      for (int j = 0; j < 4; ++j) {
        v8f acc = (j == 0) ? a0 : (j == 1) ? a1 : (j == 2) ? a2 : a3;
        int col = (ct0 + j) * 16 + lo16;
#pragma unroll
        for (int r = 0; r < 8; ++r) {
          int row = 8 * hi + r;
          int mv = mask[((size_t)(b * SS + rb * 16 + row)) * DD + col];
          sc[row * 1024 + col] = acc[r] + (float)mv * NEGV;
        }
      }
    }
  }
  __syncthreads();

  // Phase 2: row-wise softmax in LDS, emit bf16 'a' tile
  {
    int row = t >> 3, seg = t & 7;
    float* p = sc + row * 1024 + seg * 128;
    float m = -3.4e38f;
    for (int i = 0; i < 128; ++i) m = fmaxf(m, p[i]);
    red[row * 8 + seg] = m;
    __syncthreads();
    float rm = red[row * 8];
#pragma unroll
    for (int j = 1; j < 8; ++j) rm = fmaxf(rm, red[row * 8 + j]);
    __syncthreads();
    float s = 0.f;
    for (int i = 0; i < 128; ++i) { float e = __expf(p[i] - rm); p[i] = e; s += e; }
    red[row * 8 + seg] = s;
    __syncthreads();
    float rs = 0.f;
#pragma unroll
    for (int j = 0; j < 8; ++j) rs += red[row * 8 + j];
    float inv = 1.0f / rs;
    bf16_t* q = ab + row * 1024 + seg * 128;
    for (int i = 0; i < 128; ++i) q[i] = (bf16_t)(p[i] * inv);
  }
  __syncthreads();

  // Phase 3: out = a @ x_e  (B operand from transposed x copy)
  {
    const bf16_t* arow = ab + lo16 * 1024;
    for (int g = 0; g < 4; ++g) {
      int nt0 = g * 16 + wave * 4;
      const bf16_t* bc0 = xtbf + ((size_t)b * DD + nt0 * 16 + lo16) * SS;
      v8f a0 = {}, a1 = {}, a2 = {}, a3 = {};
      for (int k0 = 0; k0 < SS; k0 += 32) {
        v16bf af = load_a_frag(arow + k0, hi);
        v16bf b0 = load_b_frag(bc0 + k0, hi);
        v16bf b1 = load_b_frag(bc0 + 16 * SS + k0, hi);
        v16bf b2 = load_b_frag(bc0 + 32 * SS + k0, hi);
        v16bf b3 = load_b_frag(bc0 + 48 * SS + k0, hi);
        __builtin_prefetch(bc0 + k0 + 256, 0, 1);
        a0 = wmma_bf16(af, b0, a0);
        a1 = wmma_bf16(af, b1, a1);
        a2 = wmma_bf16(af, b2, a2);
        a3 = wmma_bf16(af, b3, a3);
      }
#pragma unroll
      for (int j = 0; j < 4; ++j) {
        v8f acc = (j == 0) ? a0 : (j == 1) ? a1 : (j == 2) ? a2 : a3;
        int col = (nt0 + j) * 16 + lo16;
#pragma unroll
        for (int r = 0; r < 8; ++r) {
          int row = 8 * hi + r;
          out[((size_t)(b * SS + rb * 16 + row)) * DD + col] = acc[r];
        }
      }
    }
  }
}

// ---------------- launcher ----------------
extern "C" void kernel_launch(void* const* d_in, const int* in_sizes, int n_in,
                              void* d_out, int out_size, void* d_ws, size_t ws_size,
                              hipStream_t stream) {
  const int*   x    = (const int*)d_in[0];
  const int*   yc   = (const int*)d_in[1];
  const int*   mask = (const int*)d_in[2];
  // d_in[3] = training (unused in inference path)
  const float* Femb = (const float*)d_in[4];
  const float* Gemb = (const float*)d_in[5];
  const float* Pw   = (const float*)d_in[6];
  const float* Pb   = (const float*)d_in[7];
  float* out = (float*)d_out;

  char* ws = (char*)d_ws;
  bf16_t* xbf  = (bf16_t*)(ws);                 // 32 MB  [b][s][d]
  bf16_t* xtbf = (bf16_t*)(ws + 33554432ull);   // 32 MB  [b][d][s]
  bf16_t* ycr  = (bf16_t*)(ws + 67108864ull);   // 128 MB [b][1024][4096]
  bf16_t* pwt  = (bf16_t*)(ws + 201326592ull);  // 8 MB   [1024][4096]
  bf16_t* mt   = (bf16_t*)(ws + 209715200ull);  // 32 MB  p_y transposed [b][j][i]

  gather_x_kernel <<<(BB * SS * DD / 8) / 256, 256, 0, stream>>>(x, Femb, xbf, xtbf);
  gather_yc_kernel<<<(BB * 4096 * DD / 8) / 256, 256, 0, stream>>>(yc, Gemb, ycr);
  pwt_kernel      <<<(DD * CC / 8) / 256, 256, 0, stream>>>(Pw, pwt);
  pygemm_kernel   <<<BB * 64, 128, 16 * CC * 2, stream>>>(ycr, pwt, Pb, mt);
  attn_kernel     <<<BB * 64, 128, 32768 + 65536 + 32768 + 512, stream>>>(xbf, xtbf, mt, mask, out);
}
